// EDUSpeakerAwareMLM_56384330662224
// MI455X (gfx1250) — matrix operand order repaired and checked by reading, not compiled, not run on hardware
//
#include <hip/hip_runtime.h>
#include <hip/hip_bf16.h>
#include <stdint.h>

// ---------------------------------------------------------------------------
// Problem constants (from the reference)
// ---------------------------------------------------------------------------
#define BB 4
#define TT 32
#define LL 64
#define DD 768
#define NHEAD 12
#define DH 64
#define DFF 2048
#define NLAYERS 2

#define NSEQ (BB * TT)          // 128 sequences entering the transformer
#define MROWS (NSEQ * LL)       // 8192 rows for all big GEMMs
#define QKVN (3 * DD)           // 2304

// ---------------------------------------------------------------------------
// WMMA types (gfx1250, wave32)
// ---------------------------------------------------------------------------
typedef __attribute__((ext_vector_type(16))) __bf16 bf16x16;
typedef __attribute__((ext_vector_type(8)))  __bf16 bf16x8;
typedef __attribute__((ext_vector_type(8)))  float  f32x8;

union BFrag { bf16x16 v; bf16x8 h[2]; };

// Load 8 consecutive f32, convert to bf16 into fragment positions [o, o+8).
__device__ __forceinline__ void cvt8(bf16x16& f, int o, const float* __restrict__ p) {
  float4 x0 = *reinterpret_cast<const float4*>(p);
  float4 x1 = *reinterpret_cast<const float4*>(p + 4);
  f[o + 0] = (__bf16)x0.x; f[o + 1] = (__bf16)x0.y;
  f[o + 2] = (__bf16)x0.z; f[o + 3] = (__bf16)x0.w;
  f[o + 4] = (__bf16)x1.x; f[o + 5] = (__bf16)x1.y;
  f[o + 6] = (__bf16)x1.z; f[o + 7] = (__bf16)x1.w;
}

// ---------------------------------------------------------------------------
// One-time f32 -> bf16 conversion for static weights.
// ---------------------------------------------------------------------------
__global__ __launch_bounds__(256) void cvt_bf16_kernel(
    const float* __restrict__ src, __bf16* __restrict__ dst, int n) {
  int i = (blockIdx.x * 256 + threadIdx.x) * 4;
  if (i + 3 < n) {
    float4 x = *reinterpret_cast<const float4*>(src + i);
    dst[i + 0] = (__bf16)x.x; dst[i + 1] = (__bf16)x.y;
    dst[i + 2] = (__bf16)x.z; dst[i + 3] = (__bf16)x.w;
  }
}

// ---------------------------------------------------------------------------
// GEMM:  C[M,N] = act(A[M,K](f32) @ W[N,K](bf16)^T + bias[N]) (+ res[M,N])
// bf16 WMMA, f32 accumulate.  Block = 128 threads = 4 waves.
// Each wave computes a 32x128 tile: 2 A-frags x 8 B-frags = 16 wmma / K-step.
// B fragments are loaded in batches of 4 so the wmma runs stay free of
// per-pair WAR hazard NOPs.  Requires M%128==0, N%128==0, K%32==0.
// ---------------------------------------------------------------------------
template <bool RELU, bool RES>
__global__ __launch_bounds__(128) void gemm_bf16_wmma(
    const float* __restrict__ A, const __bf16* __restrict__ W,
    const float* __restrict__ bias, const float* __restrict__ res,
    float* __restrict__ C, int M, int N, int K) {
  const int lane = threadIdx.x & 31;
  const int wave = threadIdx.x >> 5;
  const int hl   = lane >> 4;      // half-wave select (0: lanes 0-15, 1: 16-31)
  const int lr   = lane & 15;

  const int m0 = blockIdx.y * 128 + wave * 32;
  const int n0 = blockIdx.x * 128;

  const float* arow0 = A + (size_t)(m0 + lr) * K;
  const float* arow1 = A + (size_t)(m0 + 16 + lr) * K;

  f32x8 acc[2][8] = {};

  for (int kb = 0; kb < K; kb += 32) {
    // A fragments (16x32 bf16 each): lane row = m, K = {kb+g*16+hl*8 .. +8}
    bf16x16 a0, a1;
    cvt8(a0, 0, arow0 + kb + hl * 8);
    cvt8(a0, 8, arow0 + kb + 16 + hl * 8);
    cvt8(a1, 0, arow1 + kb + hl * 8);
    cvt8(a1, 8, arow1 + kb + 16 + hl * 8);
    if (kb + 32 < K) {            // hint next K slab (global_prefetch_b8)
      __builtin_prefetch(arow0 + kb + 32, 0, 1);
      __builtin_prefetch(arow1 + kb + 32, 0, 1);
    }
#pragma unroll
    for (int hb = 0; hb < 2; ++hb) {
      // Batch-load 4 B fragments (native bf16, 2 x b128 each), then 8 wmma.
      BFrag b[4];
#pragma unroll
      for (int j = 0; j < 4; ++j) {
        const int t = hb * 4 + j;
        const __bf16* wrow = W + (size_t)(n0 + t * 16 + lr) * K + kb + hl * 8;
        b[j].h[0] = *reinterpret_cast<const bf16x8*>(wrow);
        b[j].h[1] = *reinterpret_cast<const bf16x8*>(wrow + 16);
      }
#pragma unroll
      for (int j = 0; j < 4; ++j) {
        const int t = hb * 4 + j;
        acc[0][t] = __builtin_amdgcn_wmma_f32_16x16x32_bf16(
            false, a0, false, b[j].v, (short)0, acc[0][t], false, false);
        acc[1][t] = __builtin_amdgcn_wmma_f32_16x16x32_bf16(
            false, a1, false, b[j].v, (short)0, acc[1][t], false, false);
      }
    }
  }

  // Epilogue: C/D layout -> row = r + 8*hl, col = lane&15 (per ISA 7.12.2)
#pragma unroll
  for (int mt = 0; mt < 2; ++mt) {
#pragma unroll
    for (int t = 0; t < 8; ++t) {
      const int col = n0 + t * 16 + lr;
      const float bv = bias ? bias[col] : 0.0f;
#pragma unroll
      for (int r = 0; r < 8; ++r) {
        const int row = m0 + mt * 16 + hl * 8 + r;
        float v = acc[mt][t][r] + bv;
        if (RELU) v = fmaxf(v, 0.0f);
        if (RES)  v += res[(size_t)row * N + col];
        C[(size_t)row * N + col] = v;
      }
    }
  }
}

// ---------------------------------------------------------------------------
// edu[b,t,d] = mean_l TE[b,t,l,d]
// ---------------------------------------------------------------------------
__global__ __launch_bounds__(256) void edu_mean_kernel(
    const float* __restrict__ te, float* __restrict__ edu) {
  int idx = blockIdx.x * 256 + threadIdx.x;
  if (idx >= NSEQ * DD) return;
  int bt = idx / DD, d = idx - bt * DD;
  const float* p = te + (size_t)bt * LL * DD + d;
  float s = 0.f;
#pragma unroll 8
  for (int l = 0; l < LL; ++l) s += p[(size_t)l * DD];
  edu[idx] = s * (1.0f / LL);
}

// ---------------------------------------------------------------------------
// SAUTE context (linear attention form, avoids the 302MB M tensor):
//   x[b,t,l,:] = tok[b,t,l,:] + sum_{u<=t, spk match} (v_u . tok[b,t,l]) * k_u
// One block per (b,t).
// ---------------------------------------------------------------------------
__global__ __launch_bounds__(256) void saute_kernel(
    const float* __restrict__ tok, const float* __restrict__ kk,
    const float* __restrict__ vv, const int* __restrict__ spk,
    float* __restrict__ xout) {
  __shared__ float Ss[LL * TT];
  const int bt = blockIdx.x;          // b*TT + t
  const int b = bt >> 5, t = bt & 31;
  const int spk_t = spk[bt];
  const float* tokb = tok + (size_t)bt * LL * DD;
  const int tid = threadIdx.x;

  for (int i = tid; i < LL * TT; i += 256) {
    const int l = i >> 5, u = i & 31;
    float s = 0.f;
    if (u <= t && spk[b * TT + u] == spk_t) {
      const float* tp = tokb + (size_t)l * DD;
      const float* vp = vv + ((size_t)b * TT + u) * DD;
#pragma unroll 8
      for (int d = 0; d < DD; ++d) s += tp[d] * vp[d];
    }
    Ss[i] = s;
  }
  __syncthreads();
  for (int i = tid; i < LL * DD; i += 256) {
    const int l = i / DD, d = i - l * DD;
    float acc = tokb[i];
    const float* sp = Ss + l * TT;
#pragma unroll 8
    for (int u = 0; u < TT; ++u)
      acc += sp[u] * kk[((size_t)b * TT + u) * DD + d];
    xout[(size_t)bt * LL * DD + i] = acc;
  }
}

// ---------------------------------------------------------------------------
// Softmax attention core over qkv buffer: one block per (seq n, head h).
// qkv row layout: [q(768) | k(768) | v(768)], ld = 2304.
// Q/K tiles staged to LDS with CDNA5 async copies (ASYNCcnt protocol).
// ---------------------------------------------------------------------------
__global__ __launch_bounds__(128) void attn_kernel(
    const float* __restrict__ qkv, float* __restrict__ out) {
  __shared__ float Qs[LL * DH];
  __shared__ float Ks[LL * DH];
  __shared__ float Ss[LL * LL];
  const int n = blockIdx.x / NHEAD;
  const int h = blockIdx.x % NHEAD;
  const int tid = threadIdx.x;
  const size_t rowbase = (size_t)n * LL * QKVN;
  const int qoff = h * DH, koff = DD + h * DH, voff = 2 * DD + h * DH;

  // Async global->LDS staging: 16B per lane per issue, per-lane LDS offset in
  // VDST, 64-bit global address in VADDR (GV mode).  Tracked by ASYNCcnt.
  for (int i = tid * 4; i < LL * DH; i += 128 * 4) {
    const int l = i >> 6, d = i & 63;
    const float* gq = qkv + rowbase + (size_t)l * QKVN + qoff + d;
    const float* gk = qkv + rowbase + (size_t)l * QKVN + koff + d;
    uint32_t lq = (uint32_t)(uintptr_t)(&Qs[i]);   // low 32 bits = LDS offset
    uint32_t lk = (uint32_t)(uintptr_t)(&Ks[i]);
    asm volatile("global_load_async_to_lds_b128 %0, %1, off"
                 :: "v"(lq), "v"((uint64_t)(uintptr_t)gq) : "memory");
    asm volatile("global_load_async_to_lds_b128 %0, %1, off"
                 :: "v"(lk), "v"((uint64_t)(uintptr_t)gk) : "memory");
  }
  asm volatile("s_wait_asynccnt 0" ::: "memory");
  __syncthreads();

  for (int i = tid; i < LL * LL; i += 128) {
    const int l = i >> 6, u = i & 63;
    const float* qp = Qs + l * DH;
    const float* kp = Ks + u * DH;
    float s = 0.f;
#pragma unroll 8
    for (int d = 0; d < DH; ++d) s += qp[d] * kp[d];
    Ss[i] = s * 0.125f;               // 1/sqrt(64)
  }
  __syncthreads();
  if (tid < LL) {
    float* sp = Ss + tid * LL;
    float mx = sp[0];
    for (int u = 1; u < LL; ++u) mx = fmaxf(mx, sp[u]);
    float sum = 0.f;
    for (int u = 0; u < LL; ++u) { float e = __expf(sp[u] - mx); sp[u] = e; sum += e; }
    const float inv = 1.0f / sum;
    for (int u = 0; u < LL; ++u) sp[u] *= inv;
  }
  __syncthreads();
  for (int i = tid; i < LL * DH; i += 128) {
    const int l = i >> 6, d = i & 63;
    const float* sp = Ss + l * LL;
    float acc = 0.f;
#pragma unroll 8
    for (int u = 0; u < LL; ++u)
      acc += sp[u] * qkv[rowbase + (size_t)u * QKVN + voff + d];
    out[((size_t)n * LL + l) * DD + h * DH + d] = acc;
  }
}

// ---------------------------------------------------------------------------
// LayerNorm over D=768: one block per row, 256 threads x 3 elements.
// ---------------------------------------------------------------------------
__global__ __launch_bounds__(256) void layernorm_kernel(
    const float* __restrict__ xin, const float* __restrict__ g,
    const float* __restrict__ bta, float* __restrict__ yout) {
  __shared__ float red[256];
  const int t = threadIdx.x;
  const float* xr = xin + (size_t)blockIdx.x * DD;
  const float v0 = xr[t], v1 = xr[t + 256], v2 = xr[t + 512];
  red[t] = v0 + v1 + v2;
  __syncthreads();
  for (int o = 128; o > 0; o >>= 1) { if (t < o) red[t] += red[t + o]; __syncthreads(); }
  const float mu = red[0] * (1.0f / DD);
  __syncthreads();
  const float d0 = v0 - mu, d1 = v1 - mu, d2 = v2 - mu;
  red[t] = d0 * d0 + d1 * d1 + d2 * d2;
  __syncthreads();
  for (int o = 128; o > 0; o >>= 1) { if (t < o) red[t] += red[t + o]; __syncthreads(); }
  const float rinv = rsqrtf(red[0] * (1.0f / DD) + 1e-5f);
  float* yr = yout + (size_t)blockIdx.x * DD;
  yr[t]       = d0 * rinv * g[t]       + bta[t];
  yr[t + 256] = d1 * rinv * g[t + 256] + bta[t + 256];
  yr[t + 512] = d2 * rinv * g[t + 512] + bta[t + 512];
}

// ---------------------------------------------------------------------------
// Orchestration
// ---------------------------------------------------------------------------
extern "C" void kernel_launch(void* const* d_in, const int* in_sizes, int n_in,
                              void* d_out, int out_size, void* d_ws, size_t ws_size,
                              hipStream_t stream) {
  (void)in_sizes; (void)n_in; (void)out_size; (void)ws_size;
  const float* te    = (const float*)d_in[0];
  const int*   spk   = (const int*)  d_in[1];
  const float* Wq    = (const float*)d_in[2];
  const float* Wk    = (const float*)d_in[3];
  const float* Wv    = (const float*)d_in[4];
  const float* in_w  = (const float*)d_in[5];   // (2, 2304, 768)
  const float* in_b  = (const float*)d_in[6];   // (2, 2304)
  const float* out_w = (const float*)d_in[7];   // (2, 768, 768)
  const float* out_b = (const float*)d_in[8];   // (2, 768)
  const float* ln1_g = (const float*)d_in[9];
  const float* ln1_b = (const float*)d_in[10];
  const float* ff1_w = (const float*)d_in[11];  // (2, 2048, 768)
  const float* ff1_b = (const float*)d_in[12];  // (2, 2048)
  const float* ff2_w = (const float*)d_in[13];  // (2, 768, 2048)
  const float* ff2_b = (const float*)d_in[14];  // (2, 768)
  const float* ln2_g = (const float*)d_in[15];
  const float* ln2_b = (const float*)d_in[16];
  float* outp = (float*)d_out;

  // Workspace carve-up
  float* ws   = (float*)d_ws;
  float* edu  = ws;                              // 128*768
  float* kbuf = edu  + (size_t)NSEQ * DD;        // 128*768
  float* vbuf = kbuf + (size_t)NSEQ * DD;        // 128*768
  float* tok  = vbuf + (size_t)NSEQ * DD;        // 8192*768
  float* xbuf = tok  + (size_t)MROWS * DD;       // 8192*768
  float* ybuf = xbuf + (size_t)MROWS * DD;       // 8192*768
  float* abuf = ybuf + (size_t)MROWS * DD;       // 8192*768 (attention concat)
  float* qkv  = abuf + (size_t)MROWS * DD;       // 8192*2304
  float* ffb  = qkv  + (size_t)MROWS * QKVN;     // 8192*2048
  // bf16 weight mirrors
  __bf16* Wq_h  = (__bf16*)(ffb + (size_t)MROWS * DFF);
  __bf16* Wk_h  = Wq_h  + (size_t)DD * DD;
  __bf16* Wv_h  = Wk_h  + (size_t)DD * DD;
  __bf16* inw_h = Wv_h  + (size_t)DD * DD;              // 2*2304*768
  __bf16* outw_h = inw_h + (size_t)NLAYERS * QKVN * DD; // 2*768*768
  __bf16* ff1_h = outw_h + (size_t)NLAYERS * DD * DD;   // 2*2048*768
  __bf16* ff2_h = ff1_h + (size_t)NLAYERS * DFF * DD;   // 2*768*2048

  // 0) One-time weight conversion to bf16 (all reused across 8192 rows)
  auto cvt = [&](const float* s, __bf16* d, size_t n) {
    cvt_bf16_kernel<<<(int)((n / 4 + 255) / 256), 256, 0, stream>>>(s, d, (int)n);
  };
  cvt(Wq, Wq_h, (size_t)DD * DD);
  cvt(Wk, Wk_h, (size_t)DD * DD);
  cvt(Wv, Wv_h, (size_t)DD * DD);
  cvt(in_w, inw_h, (size_t)NLAYERS * QKVN * DD);
  cvt(out_w, outw_h, (size_t)NLAYERS * DD * DD);
  cvt(ff1_w, ff1_h, (size_t)NLAYERS * DFF * DD);
  cvt(ff2_w, ff2_h, (size_t)NLAYERS * DD * DFF);

  // 1) edu = mean_l TE
  edu_mean_kernel<<<(NSEQ * DD + 255) / 256, 256, 0, stream>>>(te, edu);

  // 2) tok = TE @ Wq^T   (8192 x 768 x 768)
  gemm_bf16_wmma<false, false><<<dim3(DD / 128, MROWS / 128), 128, 0, stream>>>(
      te, Wq_h, nullptr, nullptr, tok, MROWS, DD, DD);

  // 3) k = edu @ Wk^T, v = edu @ Wv^T  (128 x 768 x 768)
  gemm_bf16_wmma<false, false><<<dim3(DD / 128, NSEQ / 128), 128, 0, stream>>>(
      edu, Wk_h, nullptr, nullptr, kbuf, NSEQ, DD, DD);
  gemm_bf16_wmma<false, false><<<dim3(DD / 128, NSEQ / 128), 128, 0, stream>>>(
      edu, Wv_h, nullptr, nullptr, vbuf, NSEQ, DD, DD);

  // 4) SAUTE speaker-state context -> xbuf
  saute_kernel<<<NSEQ, 256, 0, stream>>>(tok, kbuf, vbuf, spk, xbuf);

  // 5) Two post-norm transformer layers
  for (int l = 0; l < NLAYERS; ++l) {
    // qkv = x @ in_proj_w^T + in_proj_b   (8192 x 2304 x 768)
    gemm_bf16_wmma<false, false><<<dim3(QKVN / 128, MROWS / 128), 128, 0, stream>>>(
        xbuf, inw_h + (size_t)l * QKVN * DD, in_b + (size_t)l * QKVN,
        nullptr, qkv, MROWS, QKVN, DD);
    // softmax attention per (seq, head)
    attn_kernel<<<NSEQ * NHEAD, 128, 0, stream>>>(qkv, abuf);
    // y = attn @ out_w^T + out_b + x      (8192 x 768 x 768)
    gemm_bf16_wmma<false, true><<<dim3(DD / 128, MROWS / 128), 128, 0, stream>>>(
        abuf, outw_h + (size_t)l * DD * DD, out_b + (size_t)l * DD,
        xbuf, ybuf, MROWS, DD, DD);
    // x = LN1(y)
    layernorm_kernel<<<MROWS, 256, 0, stream>>>(
        ybuf, ln1_g + (size_t)l * DD, ln1_b + (size_t)l * DD, xbuf);
    // ff = relu(x @ ff1^T + b1)           (8192 x 2048 x 768)
    gemm_bf16_wmma<true, false><<<dim3(DFF / 128, MROWS / 128), 128, 0, stream>>>(
        xbuf, ff1_h + (size_t)l * DFF * DD, ff1_b + (size_t)l * DFF,
        nullptr, ffb, MROWS, DFF, DD);
    // y = ff @ ff2^T + b2 + x             (8192 x 768 x 2048)
    gemm_bf16_wmma<false, true><<<dim3(DD / 128, MROWS / 128), 128, 0, stream>>>(
        ffb, ff2_h + (size_t)l * DD * DFF, ff2_b + (size_t)l * DD,
        xbuf, ybuf, MROWS, DD, DFF);
    // x = LN2(y) ; final layer writes d_out
    float* dst = (l == NLAYERS - 1) ? outp : xbuf;
    layernorm_kernel<<<MROWS, 256, 0, stream>>>(
        ybuf, ln2_g + (size_t)l * DD, ln2_b + (size_t)l * DD, dst);
  }
}
